// TACO_hashed_41686952575321
// MI455X (gfx1250) — matrix operand order repaired and checked by reading, not compiled, not run on hardware
//
#include <hip/hip_runtime.h>
#include <hip/hip_bf16.h>
#include <stdint.h>

// ---------------- types ----------------
typedef _Float16 h8  __attribute__((ext_vector_type(8)));
typedef _Float16 h16 __attribute__((ext_vector_type(16)));
typedef float    f8  __attribute__((ext_vector_type(8)));

#define NQ     262144
#define DH     256
#define MTILE  128
#define XST    264          // padded activation row stride in halfs (528 B -> 4-bank rotate/row)
#define MTST   260          // padded LDS ModTab row stride in floats (bank rotate per idx)
#define W0C    30.0f

// dynamic LDS layout (bytes)
#define XS_BYTES   (MTILE * XST * 2)          // 67584: f16 activations
#define WB_BYTES   16384                      // one 16 KB weight K-chunk
#define IDX_OFF    (XS_BYTES + 2 * WB_BYTES)
#define LV_OFF     (IDX_OFF + MTILE * 4 * 4)
#define MT_OFF     (LV_OFF + MTILE * 2 * 4)   // staged ModTab table: 16 rows x 260 f32
#define SMEM_BYTES (MT_OFF + 16 * MTST * 4)   // ~117 KB -> 2 WGs / WGP

// ---------------- prep 1: ModTab[j][16][256] = W0C * emb[j] @ modW[j] ----------------
__global__ __launch_bounds__(256) void prep_modtab(const float* __restrict__ emb,
                                                   const float* __restrict__ modW,
                                                   float* __restrict__ ModTab) {
    int t = blockIdx.x * 256 + threadIdx.x;      // 0 .. 4*16*256-1
    int col = t & 255;
    int row = (t >> 8) & 15;
    int j   = t >> 12;
    const float* e = emb  + (size_t)(j * 16 + row) * 16;
    const float* m = modW + (size_t)j * 16 * DH;
    float acc = 0.f;
#pragma unroll
    for (int k = 0; k < 16; ++k) acc += e[k] * m[k * DH + col];
    ModTab[t] = W0C * acc;
}

// ---------------- prep 2: Wh -> f16 WMMA-B fragments, bank-swizzled -------------------
__global__ __launch_bounds__(256) void prep_whb(const float* __restrict__ Wh,
                                                _Float16* __restrict__ WhB) {
    int t = blockIdx.x * 256 + threadIdx.x;      // 0 .. 12287
    int lane = t & 31;
    int nt   = (t >> 5) & 15;
    int kt   = (t >> 9) & 7;
    int j    = t >> 12;
    int grp   = (lane >> 3) & 1;
    int n     = nt * 16 + (lane & 15);
    int kbase = kt * 32 + ((lane < 16) ? 0 : 16);
    const float* src = Wh + (size_t)j * DH * DH;
    _Float16*    dst = WhB + (size_t)t * 16;
#pragma unroll
    for (int h = 0; h < 16; ++h)
        dst[h ^ (grp << 3)] = (_Float16)src[(size_t)(kbase + h) * DH + n];
}

// ---------------- CDNA5 async-to-LDS path (ASYNCcnt) ----------------------------------
__device__ __forceinline__ void async_copy_b128(unsigned lds_addr, const void* gaddr) {
    asm volatile("global_load_async_to_lds_b128 %0, %1, off"
                 :: "v"(lds_addr), "v"((unsigned long long)(uintptr_t)gaddr)
                 : "memory");
}
__device__ __forceinline__ void wait_asynccnt0() {
    asm volatile("s_wait_asynccnt 0x0" ::: "memory");
}
// stage one 16 KB weight chunk: 256 threads x 64 B
__device__ __forceinline__ void issue_chunk(const char* wbase, int c, unsigned ldsDst, int tid) {
    const char* g = wbase + (size_t)c * 16384 + tid * 64;
    unsigned    l = ldsDst + tid * 64;
    async_copy_b128(l,      g);
    async_copy_b128(l + 16, g + 16);
    async_copy_b128(l + 32, g + 32);
    async_copy_b128(l + 48, g + 48);
}
// stage one 16x256 f32 ModTab table into padded LDS rows (16 KB): 64 B per thread
__device__ __forceinline__ void issue_modtab(const float* mt, unsigned ldsDst, int tid) {
    int row = tid >> 4;            // 16 rows
    int off = (tid & 15) * 64;     // 16 x 64 B per row
    const char* g = (const char*)(mt + (size_t)row * DH) + off;
    unsigned    l = ldsDst + row * (MTST * 4) + off;
    async_copy_b128(l,      g);
    async_copy_b128(l + 16, g + 16);
    async_copy_b128(l + 32, g + 32);
    async_copy_b128(l + 48, g + 48);
}

// ---------------- fused SIREN kernel ----------------
__global__ __launch_bounds__(256) void siren_fused(const int*   __restrict__ coords,
                                                   const float* __restrict__ W0,
                                                   const float* __restrict__ b0,
                                                   const float* __restrict__ bh,
                                                   const float* __restrict__ Wl,
                                                   const float* __restrict__ bl,
                                                   const float* __restrict__ ModTab,
                                                   const _Float16* __restrict__ WhB,
                                                   float* __restrict__ out) {
    extern __shared__ char smem[];
    _Float16* Xs  = (_Float16*)smem;
    _Float16* Wb0 = (_Float16*)(smem + XS_BYTES);
    _Float16* Wb1 = (_Float16*)(smem + XS_BYTES + WB_BYTES);
    int*   IdxS = (int*)(smem + IDX_OFF);
    float* LvS  = (float*)(smem + LV_OFF);
    float* MtL  = (float*)(smem + MT_OFF);

    const int tid  = threadIdx.x;
    const int lane = tid & 31;
    const int wave = tid >> 5;
    const int mb2  = (wave & 3) * 32;   // two 16-row M blocks per wave
    const int nh   = wave >> 2;         // N half (128 cols)
    const int grp  = (lane >> 3) & 1;
    const int rowBase = blockIdx.x * MTILE;

    const unsigned ldsWb0 = (unsigned)(uintptr_t)Wb0;
    const unsigned ldsWb1 = (unsigned)(uintptr_t)Wb1;
    const unsigned ldsMt  = (unsigned)(uintptr_t)MtL;

    // kick off weight chunk 0 immediately (overlaps coord decode + layer 0)
    issue_chunk((const char*)WhB, 0, ldsWb0, tid);

    // --- per-row level decomposition: cum = {256,64,16,4,1}, dims = 4 ---
    if (tid < MTILE) {
        int cx = coords[2 * (rowBase + tid) + 0];
        int cy = coords[2 * (rowBase + tid) + 1];
        LvS[tid * 2 + 0] = (float)((cx >> 8) & 3);
        LvS[tid * 2 + 1] = (float)((cy >> 8) & 3);
        IdxS[tid * 4 + 0] = (((cy >> 6) & 3) << 2) | ((cx >> 6) & 3);
        IdxS[tid * 4 + 1] = (((cy >> 4) & 3) << 2) | ((cx >> 4) & 3);
        IdxS[tid * 4 + 2] = (((cy >> 2) & 3) << 2) | ((cx >> 2) & 3);
        IdxS[tid * 4 + 3] = (((cy     ) & 3) << 2) | ((cx     ) & 3);
    }
    __syncthreads();

    // --- layer 0: x = sin(W0C*(lv0 @ W0 + b0 + ModTab[0][idx0])) (table 0 from L2) ---
    {
        const int col = tid;
        const float w0x = W0[col], w0y = W0[DH + col], bb = b0[col];
        const float* mt0 = ModTab + col;
#pragma unroll 4
        for (int r = 0; r < MTILE; ++r) {
            float pre = LvS[r * 2] * w0x + LvS[r * 2 + 1] * w0y + bb
                      + mt0[IdxS[r * 4 + 0] * DH];
            Xs[r * XST + col] = (_Float16)__sinf(W0C * pre);
        }
    }

    // --- 3 hidden layers, software-pipelined staging over 24 weight chunks ---
    for (int j = 0; j < 3; ++j) {
        f8 acc[2][8] = {};

#pragma unroll
        for (int kt = 0; kt < 8; ++kt) {
            const int c   = j * 8 + kt;
            const int cur = kt & 1;               // j*8 even -> compile-time buffer select
            wait_asynccnt0();     // chunk c (and older stages) resident; in-order completion
            __syncthreads();      // all waves loaded; all done reading buffer cur^1
            if (kt < 7 || j < 2) issue_chunk((const char*)WhB, c + 1, cur ? ldsWb0 : ldsWb1, tid);
            if (kt == 0)          // stage ModTab table j+1 for this layer's epilogue
                issue_modtab(ModTab + (size_t)(j + 1) * 16 * DH, ldsMt, tid);
            if (c + 2 < 24)
                __builtin_prefetch((const char*)WhB + (size_t)(c + 2) * 16384 + tid * 64, 0, 1);
            const _Float16* W = cur ? Wb1 : Wb0;

            // A fragments (16x32 f16): lane<16 -> K {0..15}, lane>=16 -> K {16..31}
            const int kb = kt * 32 + ((lane < 16) ? 0 : 16);
            h16 a[2];
#pragma unroll
            for (int mi = 0; mi < 2; ++mi) {
                const int arow = mb2 + mi * 16 + (lane & 15);
                h8 alo = *(const h8*)&Xs[arow * XST + kb];
                h8 ahi = *(const h8*)&Xs[arow * XST + kb + 8];
                a[mi] = __builtin_shufflevector(alo, ahi,
                            0,1,2,3,4,5,6,7,8,9,10,11,12,13,14,15);
            }

            // B stream: one-deep software pipeline so ds_load of nt+1 overlaps WMMAs of nt
            h8 blo = *(const h8*)&W[((nh * 8) * 32 + lane) * 16 + (grp << 3)];
            h8 bhi = *(const h8*)&W[((nh * 8) * 32 + lane) * 16 + ((grp << 3) ^ 8)];
#pragma unroll
            for (int nt = 0; nt < 8; ++nt) {
                h16 b = __builtin_shufflevector(blo, bhi,
                            0,1,2,3,4,5,6,7,8,9,10,11,12,13,14,15);
                if (nt < 7) {
                    const int sb = ((nh * 8 + nt + 1) * 32 + lane) * 16;
                    blo = *(const h8*)&W[sb + (grp << 3)];        // swizzled halves:
                    bhi = *(const h8*)&W[sb + ((grp << 3) ^ 8)];  // conflict-free b128s
                }
                acc[0][nt] = __builtin_amdgcn_wmma_f32_16x16x32_f16(
                                 false, a[0], false, b, (short)0, acc[0][nt], false, false);
                acc[1][nt] = __builtin_amdgcn_wmma_f32_16x16x32_f16(
                                 false, a[1], false, b, (short)0, acc[1][nt], false, false);
            }
        }
        __syncthreads();  // all waves done reading Xs; staged ModTab table resident

        // epilogue: x = sin(W0C*(acc + bh[j] + MtL[idx_{j+1}][col])) -> f16 LDS
        // (overlaps in time with async load of next layer's first weight chunk)
        const float* bhj = bh + (size_t)j * DH;
#pragma unroll
        for (int mi = 0; mi < 2; ++mi) {
#pragma unroll
            for (int nt = 0; nt < 8; ++nt) {
                const int col = nh * 128 + nt * 16 + (lane & 15);
                const float bb = bhj[col];
#pragma unroll
                for (int m = 0; m < 8; ++m) {
                    const int row = mb2 + mi * 16 + ((lane < 16) ? m : 8 + m);
                    float pre = acc[mi][nt][m] + bb
                              + MtL[IdxS[row * 4 + (j + 1)] * MTST + col];
                    Xs[row * XST + col] = (_Float16)__sinf(W0C * pre);
                }
            }
        }
    }
    __syncthreads();

    // --- last layer: out[row][0..2] = x @ Wl + bl (Wl uniform -> scalar loads) ---
    if (tid < MTILE) {
        float a0 = bl[0], a1 = bl[1], a2 = bl[2];
        const _Float16* xr = &Xs[tid * XST];
#pragma unroll 4
        for (int k8 = 0; k8 < DH / 8; ++k8) {
            h8 xv = *(const h8*)&xr[k8 * 8];
#pragma unroll
            for (int i = 0; i < 8; ++i) {
                float x = (float)xv[i];
                int k = k8 * 8 + i;
                a0 += x * Wl[k * 3 + 0];
                a1 += x * Wl[k * 3 + 1];
                a2 += x * Wl[k * 3 + 2];
            }
        }
        float* o = out + (size_t)(rowBase + tid) * 3;
        o[0] = a0; o[1] = a1; o[2] = a2;
    }
}

// ---------------- launch ----------------
extern "C" void kernel_launch(void* const* d_in, const int* in_sizes, int n_in,
                              void* d_out, int out_size, void* d_ws, size_t ws_size,
                              hipStream_t stream) {
    (void)in_sizes; (void)n_in; (void)out_size; (void)ws_size;
    const int*   coords = (const int*)  d_in[0];
    const float* W0     = (const float*)d_in[1];
    const float* b0     = (const float*)d_in[2];
    const float* Wh     = (const float*)d_in[3];
    const float* bh     = (const float*)d_in[4];
    const float* emb    = (const float*)d_in[5];
    const float* modW   = (const float*)d_in[6];
    const float* Wl     = (const float*)d_in[7];
    const float* bl     = (const float*)d_in[8];
    float* out = (float*)d_out;

    float*    ModTab = (float*)d_ws;                         // 64 KB
    _Float16* WhB    = (_Float16*)((char*)d_ws + 65536);     // 384 KB (24 x 16 KB chunks)

    prep_modtab<<<64, 256, 0, stream>>>(emb, modW, ModTab);
    prep_whb  <<<48, 256, 0, stream>>>(Wh, WhB);
    siren_fused<<<NQ / MTILE, 256, SMEM_BYTES, stream>>>(coords, W0, b0, bh, Wl, bl,
                                                         ModTab, WhB, out);
}
